// GATLayer_23630910062807
// MI455X (gfx1250) — matrix operand order repaired and checked by reading, not compiled, not run on hardware
//
#include <hip/hip_runtime.h>
#include <math.h>

typedef __bf16 bf16_t;
typedef __attribute__((ext_vector_type(16))) __bf16 v16bf;
typedef __attribute__((ext_vector_type(8)))  float  v8f;
typedef __attribute__((ext_vector_type(4)))  int    v4i;

constexpr int B    = 4;
constexpr int N    = 2048;
constexpr int FIN  = 128;
constexpr int FOUT = 64;
constexpr int H    = 4;
constexpr float ALPHA_C = 0.2f;
constexpr float LOG2E_C = 1.44269504088896340736f;

// LDS tile: 64 f-rows x 64B (32 bf16 m-chunk), padded to 80B so the per-lane
// 2x ds_load_b128 B-tile reads start on 16 distinct banks.
#define TILE_STRIDE 80
#define TILE_BYTES  (FOUT * TILE_STRIDE)

// ---- CDNA5 async global->LDS path (ASYNCcnt), with compile-safe fallback ----
#if defined(__HIP_DEVICE_COMPILE__) && __has_builtin(__builtin_amdgcn_global_load_async_to_lds_b128)
#define HAVE_ASYNC 1
typedef __attribute__((address_space(1))) v4i g_v4i;
typedef __attribute__((address_space(3))) v4i l_v4i;
#else
#define HAVE_ASYNC 0
#endif

__device__ __forceinline__ void copy16_g2lds(const void* g, void* l) {
#if HAVE_ASYNC
  __builtin_amdgcn_global_load_async_to_lds_b128((g_v4i*)g, (l_v4i*)l, 0, 0);
#else
  *(uint4*)l = *(const uint4*)g;   // sync staging fallback (ds_store_b128)
#endif
}

__device__ __forceinline__ void async_wait0() {
#if HAVE_ASYNC
  asm volatile("s_wait_asynccnt 0" ::: "memory");
#endif
}

// ---------------------------------------------------------------------------
// Kernel 1: h_prime = h[b](2048x128) * W[h](128x64), split-bf16 WMMA
// (hi+lo, 3 WMMAs/tile -> near-fp32). Emits e_i/e_j PRE-SCALED by log2(e)
// (leaky_relu is positively homogeneous, so softmax is exactly preserved and
// k2 can use v_exp_f32 directly), plus transposed bf16 hT[b][h][f][m] in the
// WMMA B-matrix feed order.
// ---------------------------------------------------------------------------
__global__ __launch_bounds__(32) void k1_hprime(
    const float* __restrict__ h, const float* __restrict__ W,
    const float* __restrict__ av,
    bf16_t* __restrict__ hT, float* __restrict__ ei, float* __restrict__ ej)
{
  const int lane = threadIdx.x;
  const int col  = lane & 15;   // B/C: N index; A: row M
  const int half = lane >> 4;   // 0/1
  const int n0 = blockIdx.x * 16;
  const int hh = blockIdx.y;
  const int b  = blockIdx.z;

  v8f acc[4] = {};

  const float* hrow = h + ((size_t)b * N + (n0 + col)) * FIN;
  const float* Wh   = W + (size_t)hh * FIN * FOUT;

  #pragma unroll
  for (int kt = 0; kt < 4; ++kt) {
    const int k0 = kt * 32;
    float4 r0 = *(const float4*)(hrow + k0 + half * 16);
    float4 r1 = *(const float4*)(hrow + k0 + half * 16 + 4);
    float4 r2 = *(const float4*)(hrow + k0 + half * 16 + 8);
    float4 r3 = *(const float4*)(hrow + k0 + half * 16 + 12);
    float af[16] = {r0.x, r0.y, r0.z, r0.w, r1.x, r1.y, r1.z, r1.w,
                    r2.x, r2.y, r2.z, r2.w, r3.x, r3.y, r3.z, r3.w};
    v16bf Ahi, Alo;
    #pragma unroll
    for (int e = 0; e < 16; ++e) {
      float x = af[e];
      bf16_t hi = (bf16_t)x;
      Ahi[e] = hi;
      Alo[e] = (bf16_t)(x - (float)hi);
    }
    #pragma unroll
    for (int ft = 0; ft < 4; ++ft) {
      const float* wc = Wh + (size_t)(k0 + half * 16) * FOUT + ft * 16 + col;
      v16bf Bhi, Blo;
      #pragma unroll
      for (int j = 0; j < 16; ++j) {
        float x = wc[(size_t)j * FOUT];
        bf16_t hi = (bf16_t)x;
        Bhi[j] = hi;
        Blo[j] = (bf16_t)(x - (float)hi);
      }
      acc[ft] = __builtin_amdgcn_wmma_f32_16x16x32_bf16(false, Ahi, false, Bhi,
                                                        (short)0, acc[ft], false, false);
      acc[ft] = __builtin_amdgcn_wmma_f32_16x16x32_bf16(false, Ahi, false, Blo,
                                                        (short)0, acc[ft], false, false);
      acc[ft] = __builtin_amdgcn_wmma_f32_16x16x32_bf16(false, Alo, false, Bhi,
                                                        (short)0, acc[ft], false, false);
    }
  }

  const int bh = b * H + hh;
  bf16_t* hTb = hT + (size_t)bh * FOUT * N;
  float pei[8], pej[8];
  #pragma unroll
  for (int r = 0; r < 8; ++r) { pei[r] = 0.f; pej[r] = 0.f; }

  #pragma unroll
  for (int ft = 0; ft < 4; ++ft) {
    const int f = ft * 16 + col;
    const float a1v = av[(size_t)hh * 2 * FOUT + f];
    const float a2v = av[(size_t)hh * 2 * FOUT + FOUT + f];
    #pragma unroll
    for (int r = 0; r < 8; ++r) {
      float v = acc[ft][r];
      pei[r] += v * a1v;
      pej[r] += v * a2v;
      hTb[(size_t)f * N + (n0 + half * 8 + r)] = (bf16_t)v;
    }
  }
  #pragma unroll
  for (int m = 8; m >= 1; m >>= 1) {
    #pragma unroll
    for (int r = 0; r < 8; ++r) {
      pei[r] += __shfl_xor(pei[r], m, 32);
      pej[r] += __shfl_xor(pej[r], m, 32);
    }
  }
  if (col == 0) {
    #pragma unroll
    for (int r = 0; r < 8; ++r) {
      ei[(size_t)bh * N + n0 + half * 8 + r] = pei[r] * LOG2E_C;  // log2-domain
      ej[(size_t)bh * N + n0 + half * 8 + r] = pej[r] * LOG2E_C;
    }
  }
}

// ---------------------------------------------------------------------------
// per-(b,h) max of e_j (exact softmax row max = leaky(ei[n] + max_m ej[m]))
// ---------------------------------------------------------------------------
__global__ __launch_bounds__(256) void k_maxej(const float* __restrict__ ej,
                                               float* __restrict__ Mj)
{
  __shared__ float s[256];
  const int bh = blockIdx.x;
  const int t  = threadIdx.x;
  float m = -3.4e38f;
  for (int i = t; i < N; i += 256) m = fmaxf(m, ej[(size_t)bh * N + i]);
  s[t] = m;
  __syncthreads();
  for (int off = 128; off > 0; off >>= 1) {
    if (t < off) s[t] = fmaxf(s[t], s[t + off]);
    __syncthreads();
  }
  if (t == 0) Mj[bh] = s[0];
}

// ---------------------------------------------------------------------------
// Kernel 2: fused attention. 256-thread block = 8 waves = 128 rows per (b,h)
// block; the 32x64 bf16 h_prime K-chunk (4KB) is staged once per block into
// a double-buffered LDS tile via global_load_async_to_lds_b128 (prefetch
// chunk s+1 while computing s), cutting global/L2 B-operand traffic 8x.
// Each wave: builds its P tile in registers (16 exp2 per lane per chunk, in
// contiguous k = 16*half + e A order, same as B), 2x ds_load_b128 per B tile, 4 WMMAs per chunk.
// ---------------------------------------------------------------------------
__global__ __launch_bounds__(256) void k2_attn(
    const bf16_t* __restrict__ hT, const float* __restrict__ ei,
    const float* __restrict__ ej, const float* __restrict__ Mj,
    float* __restrict__ hnew)
{
  __shared__ __align__(16) unsigned char tile[2][TILE_BYTES];

  const int tid  = threadIdx.x;
  const int lane = tid & 31;
  const int wv   = tid >> 5;          // wave id 0..7
  const int col  = lane & 15;
  const int half = (lane >> 4) & 1;
  const int n0 = blockIdx.x * 128 + wv * 16;
  const int bh = blockIdx.z * H + blockIdx.y;

  const float eiv = ei[(size_t)bh * N + n0 + col];   // lane's row (M = col)
  const float mj  = Mj[bh];
  float tmax = eiv + mj;
  const float rowmax = tmax > 0.f ? tmax : ALPHA_C * tmax;

  const float* ejb = ej + (size_t)bh * N;
  const char*  hTb = (const char*)(hT + (size_t)bh * FOUT * N);

  // staging assignment: 4KB chunk = 256 x b128, exactly one per thread
  const int    cf   = tid >> 2;                       // f row 0..63
  const int    cp   = tid & 3;                        // 16B part
  const size_t goff = (size_t)cf * (N * 2) + (size_t)cp * 16;
  const int    loff = cf * TILE_STRIDE + cp * 16;

  copy16_g2lds(hTb + goff, &tile[0][loff]);           // prefetch chunk 0

  v8f acc[4] = {};
  float dpart = 0.f;

  for (int s = 0; s < N / 32; ++s) {
    const int m0 = s * 32;
    async_wait0();            // this wave's async loads for tile[s&1] done
    __syncthreads();          // all waves loaded tile[s&1]; prev reads retired
    if (s + 1 < N / 32)
      copy16_g2lds(hTb + goff + (size_t)(m0 + 32) * 2, &tile[(s + 1) & 1][loff]);

    // P tile: lane computes exactly the 16 probabilities its A slot needs
    float4 r0 = *(const float4*)(ejb + m0 + half * 16);
    float4 r1 = *(const float4*)(ejb + m0 + half * 16 + 4);
    float4 r2 = *(const float4*)(ejb + m0 + half * 16 + 8);
    float4 r3 = *(const float4*)(ejb + m0 + half * 16 + 12);
    float ev[16] = {r0.x, r0.y, r0.z, r0.w, r1.x, r1.y, r1.z, r1.w,
                    r2.x, r2.y, r2.z, r2.w, r3.x, r3.y, r3.z, r3.w};
    v16bf P;
    #pragma unroll
    for (int e = 0; e < 16; ++e) {
      float l = eiv + ev[e];
      l = l > 0.f ? l : ALPHA_C * l;        // leaky_relu (log2 domain)
      float p = exp2f(l - rowmax);          // v_exp_f32, == exp() of reference
      dpart += p;
      P[e] = (bf16_t)p;
    }

    const unsigned char* tb = tile[s & 1];
    #pragma unroll
    for (int ft = 0; ft < 4; ++ft) {
      union { uint4 q[2]; v16bf v; } u;
      const unsigned char* bp = tb + (ft * 16 + col) * TILE_STRIDE + half * 32;
      u.q[0] = *(const uint4*)bp;
      u.q[1] = *(const uint4*)(bp + 16);
      acc[ft] = __builtin_amdgcn_wmma_f32_16x16x32_bf16(false, P, false, u.v,
                                                        (short)0, acc[ft], false, false);
    }
  }

  // row denominator: lanes L and L+16 jointly cover all m for row L%16
  float denom = dpart + __shfl_xor(dpart, 16, 32);
  float* hb = hnew + ((size_t)bh * N + n0) * FOUT;
  #pragma unroll
  for (int r = 0; r < 8; ++r) {
    float inv = 1.f / __shfl(denom, r + half * 8, 32);
    #pragma unroll
    for (int ft = 0; ft < 4; ++ft)
      hb[(size_t)(half * 8 + r) * FOUT + ft * 16 + col] = acc[ft][r] * inv;
  }
}

// ---------------------------------------------------------------------------
// Kernel 3: mean over heads + ELU -> (B, N, FOUT) fp32
// ---------------------------------------------------------------------------
__global__ __launch_bounds__(256) void k3_out(const float* __restrict__ hnew,
                                              float* __restrict__ out)
{
  const int idx = blockIdx.x * 256 + threadIdx.x;
  if (idx >= B * N * FOUT) return;
  const int f = idx & (FOUT - 1);
  const int n = (idx >> 6) & (N - 1);
  const int b = idx >> 17;
  float s = 0.f;
  #pragma unroll
  for (int hh = 0; hh < H; ++hh)
    s += hnew[(((size_t)b * H + hh) * N + n) * FOUT + f];
  s *= (1.f / H);
  out[idx] = s > 0.f ? s : (__expf(s) - 1.f);
}

// ---------------------------------------------------------------------------
extern "C" void kernel_launch(void* const* d_in, const int* in_sizes, int n_in,
                              void* d_out, int out_size, void* d_ws, size_t ws_size,
                              hipStream_t stream) {
  (void)in_sizes; (void)n_in; (void)out_size; (void)ws_size;
  const float* h = (const float*)d_in[0];   // (B, N, FIN)
  const float* W = (const float*)d_in[1];   // (H, FIN, FOUT)
  const float* a = (const float*)d_in[2];   // (H, 2*FOUT, 1)
  float* out = (float*)d_out;               // (B, N, FOUT)

  char* ws = (char*)d_ws;
  bf16_t* hT = (bf16_t*)ws;                                   // 4 MB
  size_t off = (size_t)B * H * FOUT * N * sizeof(bf16_t);
  float* ei = (float*)(ws + off); off += (size_t)B * H * N * sizeof(float);
  float* ej = (float*)(ws + off); off += (size_t)B * H * N * sizeof(float);
  float* Mj = (float*)(ws + off); off += 256;
  float* hnew = (float*)(ws + off);                           // 8 MB

  dim3 g1(N / 16, H, B);
  k1_hprime<<<g1, 32, 0, stream>>>(h, W, a, hT, ei, ej);
  k_maxej<<<B * H, 256, 0, stream>>>(ej, Mj);
  dim3 g2(N / 128, H, B);
  k2_attn<<<g2, 256, 0, stream>>>(hT, ei, ej, Mj, hnew);
  k3_out<<<(B * N * FOUT + 255) / 256, 256, 0, stream>>>(hnew, out);
}
